// EncoderLayer_54820962566895
// MI455X (gfx1250) — matrix operand order repaired and checked by reading, not compiled, not run on hardware
//
#include <hip/hip_runtime.h>
#include <hip/hip_bf16.h>
#include <math.h>

// ---------------- types ----------------
typedef _Float16 h16;
typedef __attribute__((ext_vector_type(16))) _Float16 v16h;
typedef __attribute__((ext_vector_type(8)))  float    v8f;

#define D_MODEL 1024
#define D_FF    4096
#define SEQ     2048
#define BATCH   2
#define NHEAD   16
#define DHEAD   64
#define MTOT    (BATCH * SEQ)   // 4096 token rows

// ---------------- WMMA fragment loaders (wave32) ----------------
// A-matrix 16x32 f16 layout: lane holds row M=lane%16; K contiguous:
// lanes 0-15 -> K=0..15, lanes 16-31 -> K=16..31 (matches B frags).
__device__ __forceinline__ v16h load_a_frag(const h16* s, int ldm, int rowBase, int ksel) {
  const int lane = threadIdx.x & 31;
  const int r = rowBase + (lane & 15);
  const int kb = (lane < 16) ? 0 : 16;
  const h16* p = s + r * ldm + ksel * 32;
  v16h f;
#pragma unroll
  for (int i = 0; i < 4; ++i) {
    f[2*i]     = p[kb + 2*i];
    f[2*i + 1] = p[kb + 2*i + 1];
    f[8 + 2*i]     = p[kb + 8 + 2*i];
    f[8 + 2*i + 1] = p[kb + 8 + 2*i + 1];
  }
  return f;
}

// B-matrix 32x16 f16: lane holds column N=lane%16; K contiguous:
// lanes 0-15 -> K=0..15, lanes 16-31 -> K=16..31.
// Source stored transposed Bt[n][k] -> fully contiguous reads.
__device__ __forceinline__ v16h load_b_frag(const h16* s, int ldm, int colBase, int ksel) {
  const int lane = threadIdx.x & 31;
  const int c = colBase + (lane & 15);
  const int kb = ((lane < 16) ? 0 : 16) + ksel * 32;
  const h16* p = s + c * ldm + kb;
  v16h f;
#pragma unroll
  for (int j = 0; j < 16; ++j) f[j] = p[j];
  return f;
}

// Same B layout but source stored [k][n] (e.g. a V tile): strided reads.
__device__ __forceinline__ v16h load_bT_frag(const h16* s, int ldm, int colBase, int ksel) {
  const int lane = threadIdx.x & 31;
  const int c = colBase + (lane & 15);
  const int kb = ((lane < 16) ? 0 : 16) + ksel * 32;
  v16h f;
#pragma unroll
  for (int j = 0; j < 16; ++j) f[j] = s[(kb + j) * ldm + c];
  return f;
}

__device__ __forceinline__ v8f wmma_f16(v16h a, v16h b, v8f c) {
  return __builtin_amdgcn_wmma_f32_16x16x32_f16(false, a, false, b, (short)0, c, false, false);
}

// ---------------- conversion kernels ----------------
__global__ void cvt_f32_to_f16(const float* __restrict__ in, h16* __restrict__ out, size_t n) {
  size_t i = (size_t)blockIdx.x * blockDim.x + threadIdx.x;
  size_t stride = (size_t)gridDim.x * blockDim.x;
  for (; i < n; i += stride) out[i] = (h16)in[i];
}

// Tiled, coalesced transpose+convert: WT[n*K + k] = (h16) W[k*N + n].
// 256 threads handle a 32x32 tile through padded LDS.
__global__ __launch_bounds__(256) void transpose_cvt(
    const float* __restrict__ W, h16* __restrict__ WT, int K, int N) {
  __shared__ float t[32][33];
  const int k0 = blockIdx.x * 32;
  const int n0 = blockIdx.y * 32;
  const int tx = threadIdx.x & 31;
  const int ty = threadIdx.x >> 5;          // 0..7
#pragma unroll
  for (int i = 0; i < 4; ++i) {
    int k = ty + i * 8;
    t[k][tx] = W[(size_t)(k0 + k) * N + n0 + tx];
  }
  __syncthreads();
#pragma unroll
  for (int i = 0; i < 4; ++i) {
    int n = ty + i * 8;
    WT[(size_t)(n0 + n) * K + k0 + tx] = (h16)t[tx][n];
  }
}

// ---------------- WMMA GEMM: C = A[MxK] * Bt^T + bias, with epilogue ----------------
// Software-pipelined: tile k+1 global_load_b128 issued before tile-k WMMAs.
// EPI 0: outH = (h16)v        (QKV projections)
// EPI 1: outH = (h16)gelu(v)  (FFN up-proj, exact erf gelu)
// EPI 2: outF = res + v       (fp32 residual add: Wo / FFN down-proj)
template <int EPI>
__global__ __launch_bounds__(256) void gemm_wmma(
    const h16* __restrict__ A, const h16* __restrict__ Bt,
    const float* __restrict__ bias, const float* __restrict__ res,
    h16* __restrict__ outH, float* __restrict__ outF,
    int M, int N, int K) {
  constexpr int LDT = 34;                 // 128x32 tile, 17-bank row stride (conflict-free)
  __shared__ h16 sA[128 * LDT];
  __shared__ h16 sB[128 * LDT];

  const int tid  = threadIdx.x;
  const int lane = tid & 31;
  const int wave = tid >> 5;
  const int wm = wave & 1, wn = wave >> 1;    // 2 (M) x 4 (N) waves
  const int m0 = wm * 64, n0 = wn * 32;
  const int bM = blockIdx.y * 128, bN = blockIdx.x * 128;

  v8f acc[4][2];
  const v8f vz = {0.f,0.f,0.f,0.f,0.f,0.f,0.f,0.f};
#pragma unroll
  for (int i = 0; i < 4; ++i)
#pragma unroll
    for (int j = 0; j < 2; ++j) acc[i][j] = vz;

  // per-thread staging: 2 x uint4 per tile per matrix (128x32 halves = 512 uint4)
  uint4 ra[2], rb[2];
  const int nk = K >> 5;

  auto load_tile = [&](int kt) {
#pragma unroll
    for (int j = 0; j < 2; ++j) {
      int idx = tid + j * 256;
      int row = idx >> 2;                  // 4 uint4 per 32-half row
      int q   = idx & 3;
      ra[j] = *(const uint4*)(A  + ((size_t)(bM + row)) * K + (size_t)kt * 32 + q * 8);
      rb[j] = *(const uint4*)(Bt + ((size_t)(bN + row)) * K + (size_t)kt * 32 + q * 8);
    }
  };
  auto store_tile = [&]() {
#pragma unroll
    for (int j = 0; j < 2; ++j) {
      int idx = tid + j * 256;
      int row = idx >> 2;
      int q   = idx & 3;
      unsigned int* pA = (unsigned int*)sA + row * 17 + q * 4;  // LDT/2 = 17 dwords/row
      pA[0] = ra[j].x; pA[1] = ra[j].y; pA[2] = ra[j].z; pA[3] = ra[j].w;
      unsigned int* pB = (unsigned int*)sB + row * 17 + q * 4;
      pB[0] = rb[j].x; pB[1] = rb[j].y; pB[2] = rb[j].z; pB[3] = rb[j].w;
    }
  };

  load_tile(0);
  for (int kt = 0; kt < nk; ++kt) {
    __syncthreads();          // previous compute done reading LDS
    store_tile();
    __syncthreads();
    if (kt + 1 < nk) load_tile(kt + 1);   // overlap VMEM with WMMA below

    v16h af[4], bf[2];
#pragma unroll
    for (int i = 0; i < 4; ++i) af[i] = load_a_frag(sA, LDT, m0 + i * 16, 0);
#pragma unroll
    for (int j = 0; j < 2; ++j) bf[j] = load_b_frag(sB, LDT, n0 + j * 16, 0);
#pragma unroll
    for (int i = 0; i < 4; ++i)
#pragma unroll
      for (int j = 0; j < 2; ++j) acc[i][j] = wmma_f16(af[i], bf[j], acc[i][j]);
  }

  // epilogue: C tile layout: VGPR r -> row r (lanes 0-15) / row r+8 (lanes 16-31)
  const int hi8 = (lane >= 16) ? 8 : 0;
  const int ln  = lane & 15;
#pragma unroll
  for (int i = 0; i < 4; ++i)
#pragma unroll
    for (int j = 0; j < 2; ++j)
#pragma unroll
      for (int r = 0; r < 8; ++r) {
        int row = bM + m0 + i * 16 + r + hi8;
        int col = bN + n0 + j * 16 + ln;
        float v = acc[i][j][r] + bias[col];
        size_t o = (size_t)row * N + col;
        if (EPI == 0) {
          outH[o] = (h16)v;
        } else if (EPI == 1) {
          float g = 0.5f * v * (1.0f + erff(v * 0.70710678118654752f));
          outH[o] = (h16)g;
        } else {
          outF[o] = res[o] + v;
        }
      }
}

// ---------------- flash-style attention (per b,h, 64-query tile) ----------------
__global__ __launch_bounds__(128) void attn_kernel(
    const h16* __restrict__ Q, const h16* __restrict__ Kb,
    const h16* __restrict__ Vb, h16* __restrict__ ctx) {
  constexpr int LQ = 66;                  // 64x64 tiles, 33-bank row stride (conflict-free)
  __shared__ h16 sQ[64 * LQ];
  __shared__ h16 sK[64 * LQ];
  __shared__ h16 sV[64 * LQ];
  __shared__ h16 sP[64 * LQ];

  const int tid  = threadIdx.x;
  const int lane = tid & 31;
  const int wave = tid >> 5;              // 4 waves, each owns 16 query rows
  const int wbase = wave * 16;
  const int bh = blockIdx.y;
  const int b  = bh >> 4;
  const int h  = bh & 15;
  const int qblk = blockIdx.x;
  const size_t headOff = (size_t)h * DHEAD;

  // 64 rows x 64 halves = 512 uint4; 4 per thread, b128 global loads
  auto fill_tile = [&](h16* dst, const h16* src, int rowBase) {
#pragma unroll
    for (int j = 0; j < 4; ++j) {
      int idx = tid + j * 128;
      int row = idx >> 3;                 // 8 uint4 per 64-half row
      int q   = idx & 7;
      uint4 v = *(const uint4*)(src + ((size_t)(b * SEQ + rowBase + row)) * D_MODEL + headOff + q * 8);
      unsigned int* p = (unsigned int*)dst + row * 33 + q * 4;   // LQ/2 = 33 dwords/row
      p[0] = v.x; p[1] = v.y; p[2] = v.z; p[3] = v.w;
    }
  };

  fill_tile(sQ, Q, qblk * 64);

  float mrow[8], lrow[8];
#pragma unroll
  for (int r = 0; r < 8; ++r) { mrow[r] = -1e30f; lrow[r] = 0.0f; }
  v8f Oc[4];
  const v8f vz = {0.f,0.f,0.f,0.f,0.f,0.f,0.f,0.f};
#pragma unroll
  for (int nt = 0; nt < 4; ++nt) Oc[nt] = vz;

  const int hi8 = (lane >= 16) ? 8 : 0;
  const int ln  = lane & 15;

  for (int kb = 0; kb < SEQ / 64; ++kb) {
    __syncthreads();
    fill_tile(sK, Kb, kb * 64);
    fill_tile(sV, Vb, kb * 64);
    __syncthreads();

    // S = Q * K^T  (K tile is exactly Bt=[key][d])
    v8f S[4];
#pragma unroll
    for (int nt = 0; nt < 4; ++nt) S[nt] = vz;
#pragma unroll
    for (int ks = 0; ks < 2; ++ks) {
      v16h aq = load_a_frag(sQ, LQ, wbase, ks);
#pragma unroll
      for (int nt = 0; nt < 4; ++nt) {
        v16h bk = load_b_frag(sK, LQ, nt * 16, ks);
        S[nt] = wmma_f16(aq, bk, S[nt]);
      }
    }

    // online softmax per row; row stats live in the 16-lane group holding that row
#pragma unroll
    for (int r = 0; r < 8; ++r) {
      float vs[4];
      float mx = -1e30f;
#pragma unroll
      for (int nt = 0; nt < 4; ++nt) { vs[nt] = S[nt][r] * 0.125f; mx = fmaxf(mx, vs[nt]); }
#pragma unroll
      for (int off = 8; off >= 1; off >>= 1) mx = fmaxf(mx, __shfl_xor(mx, off, 32));
      float mnew = fmaxf(mrow[r], mx);
      float corr = __expf(mrow[r] - mnew);
      float ps = 0.0f;
#pragma unroll
      for (int nt = 0; nt < 4; ++nt) {
        float p = __expf(vs[nt] - mnew);
        ps += p;
        sP[(wbase + r + hi8) * LQ + nt * 16 + ln] = (h16)p;
      }
#pragma unroll
      for (int off = 8; off >= 1; off >>= 1) ps += __shfl_xor(ps, off, 32);
      lrow[r] = lrow[r] * corr + ps;
      mrow[r] = mnew;
#pragma unroll
      for (int nt = 0; nt < 4; ++nt) Oc[nt][r] *= corr;
    }
    __syncthreads();   // order sP writes vs. fragment reads

    // O += P * V   (V tile stored [key][d] -> transposed B reads)
#pragma unroll
    for (int ks = 0; ks < 2; ++ks) {
      v16h ap = load_a_frag(sP, LQ, wbase, ks);
#pragma unroll
      for (int nt = 0; nt < 4; ++nt) {
        v16h bv = load_bT_frag(sV, LQ, nt * 16, ks);
        Oc[nt] = wmma_f16(ap, bv, Oc[nt]);
      }
    }
  }

  // normalize and write context (merged-head layout [token][h*64+d])
#pragma unroll
  for (int nt = 0; nt < 4; ++nt)
#pragma unroll
    for (int r = 0; r < 8; ++r) {
      int q = qblk * 64 + wbase + r + hi8;
      int d = nt * 16 + ln;
      float val = Oc[nt][r] / lrow[r];
      ctx[((size_t)(b * SEQ + q)) * D_MODEL + headOff + d] = (h16)val;
    }
}

// ---------------- LayerNorm (block per row) ----------------
__global__ __launch_bounds__(256) void ln_kernel(
    const float* __restrict__ y, const float* __restrict__ g, const float* __restrict__ be,
    float* __restrict__ outF, h16* __restrict__ outH, int D) {
  __shared__ float red[256];
  const int row = blockIdx.x;
  const int tid = threadIdx.x;
  const float* p = y + (size_t)row * D;

  float s = 0.0f;
  for (int c = tid; c < D; c += 256) s += p[c];
  red[tid] = s; __syncthreads();
  for (int k = 128; k > 0; k >>= 1) { if (tid < k) red[tid] += red[tid + k]; __syncthreads(); }
  float mu = red[0] / D;
  __syncthreads();

  float vs = 0.0f;
  for (int c = tid; c < D; c += 256) { float d = p[c] - mu; vs += d * d; }
  red[tid] = vs; __syncthreads();
  for (int k = 128; k > 0; k >>= 1) { if (tid < k) red[tid] += red[tid + k]; __syncthreads(); }
  float inv = rsqrtf(red[0] / D + 1e-5f);

  for (int c = tid; c < D; c += 256) {
    float o = (p[c] - mu) * inv * g[c] + be[c];
    outF[(size_t)row * D + c] = o;
    if (outH) outH[(size_t)row * D + c] = (h16)o;
  }
}

// ---------------- host driver ----------------
extern "C" void kernel_launch(void* const* d_in, const int* in_sizes, int n_in,
                              void* d_out, int out_size, void* d_ws, size_t ws_size,
                              hipStream_t stream) {
  const float* x    = (const float*)d_in[0];
  const float* Wq   = (const float*)d_in[1];
  const float* bq   = (const float*)d_in[2];
  const float* Wk   = (const float*)d_in[3];
  const float* bk   = (const float*)d_in[4];
  const float* Wv   = (const float*)d_in[5];
  const float* bv   = (const float*)d_in[6];
  const float* Wo   = (const float*)d_in[7];
  const float* bo   = (const float*)d_in[8];
  const float* W1   = (const float*)d_in[9];
  const float* b1   = (const float*)d_in[10];
  const float* W2   = (const float*)d_in[11];
  const float* b2   = (const float*)d_in[12];
  const float* ln1g = (const float*)d_in[13];
  const float* ln1b = (const float*)d_in[14];
  const float* ln2g = (const float*)d_in[15];
  const float* ln2b = (const float*)d_in[16];
  float* out = (float*)d_out;

  // workspace carve-up
  char* base = (char*)d_ws;
  size_t off = 0;
  auto alloc = [&](size_t bytes) -> char* {
    char* p = base + off;
    off = (off + bytes + 255) & ~(size_t)255;
    return p;
  };
  const size_t MD = (size_t)MTOT * D_MODEL;
  const size_t MF = (size_t)MTOT * D_FF;
  h16*  xh   = (h16*)alloc(MD * 2);
  h16*  WqT  = (h16*)alloc((size_t)D_MODEL * D_MODEL * 2);
  h16*  WkT  = (h16*)alloc((size_t)D_MODEL * D_MODEL * 2);
  h16*  WvT  = (h16*)alloc((size_t)D_MODEL * D_MODEL * 2);
  h16*  WoT  = (h16*)alloc((size_t)D_MODEL * D_MODEL * 2);
  h16*  W1T  = (h16*)alloc((size_t)D_MODEL * D_FF * 2);
  h16*  W2T  = (h16*)alloc((size_t)D_FF * D_MODEL * 2);
  h16*  Qb   = (h16*)alloc(MD * 2);
  h16*  Kb   = (h16*)alloc(MD * 2);
  h16*  Vb   = (h16*)alloc(MD * 2);
  h16*  ctx  = (h16*)alloc(MD * 2);
  h16*  x1h  = (h16*)alloc(MD * 2);
  h16*  hbuf = (h16*)alloc(MF * 2);
  float* y1  = (float*)alloc(MD * 4);   // reused for FFN output
  float* x1  = (float*)alloc(MD * 4);

  // 1) precision conversion / weight transpose (tiled & coalesced)
  cvt_f32_to_f16<<<4096, 256, 0, stream>>>(x, xh, MD);
  {
    dim3 gT(D_MODEL / 32, D_MODEL / 32);
    transpose_cvt<<<gT, 256, 0, stream>>>(Wq, WqT, D_MODEL, D_MODEL);
    transpose_cvt<<<gT, 256, 0, stream>>>(Wk, WkT, D_MODEL, D_MODEL);
    transpose_cvt<<<gT, 256, 0, stream>>>(Wv, WvT, D_MODEL, D_MODEL);
    transpose_cvt<<<gT, 256, 0, stream>>>(Wo, WoT, D_MODEL, D_MODEL);
    dim3 gT1(D_MODEL / 32, D_FF / 32);
    transpose_cvt<<<gT1, 256, 0, stream>>>(W1, W1T, D_MODEL, D_FF);
    dim3 gT2(D_FF / 32, D_MODEL / 32);
    transpose_cvt<<<gT2, 256, 0, stream>>>(W2, W2T, D_FF, D_MODEL);
  }

  // 2) QKV projections (f16 out, head-interleaved layout preserved)
  dim3 gProj(D_MODEL / 128, MTOT / 128);
  gemm_wmma<0><<<gProj, 256, 0, stream>>>(xh, WqT, bq, nullptr, Qb, nullptr, MTOT, D_MODEL, D_MODEL);
  gemm_wmma<0><<<gProj, 256, 0, stream>>>(xh, WkT, bk, nullptr, Kb, nullptr, MTOT, D_MODEL, D_MODEL);
  gemm_wmma<0><<<gProj, 256, 0, stream>>>(xh, WvT, bv, nullptr, Vb, nullptr, MTOT, D_MODEL, D_MODEL);

  // 3) flash attention: grid = (query tiles, batch*heads)
  dim3 gAttn(SEQ / 64, BATCH * NHEAD);
  attn_kernel<<<gAttn, 128, 0, stream>>>(Qb, Kb, Vb, ctx);

  // 4) output projection + residual (fp32)
  gemm_wmma<2><<<gProj, 256, 0, stream>>>(ctx, WoT, bo, x, nullptr, y1, MTOT, D_MODEL, D_MODEL);

  // 5) LayerNorm 1 -> x1 (fp32 residual) + x1h (f16 GEMM operand)
  ln_kernel<<<MTOT, 256, 0, stream>>>(y1, ln1g, ln1b, x1, x1h, D_MODEL);

  // 6) FFN up-proj + exact gelu (f16 out)
  dim3 gFF1(D_FF / 128, MTOT / 128);
  gemm_wmma<1><<<gFF1, 256, 0, stream>>>(x1h, W1T, b1, nullptr, hbuf, nullptr, MTOT, D_FF, D_MODEL);

  // 7) FFN down-proj + residual (fp32, reuse y1)
  gemm_wmma<2><<<gProj, 256, 0, stream>>>(hbuf, W2T, b2, x1, nullptr, y1, MTOT, D_MODEL, D_FF);

  // 8) LayerNorm 2 -> final fp32 output
  ln_kernel<<<MTOT, 256, 0, stream>>>(y1, ln2g, ln2b, out, nullptr, D_MODEL);
}
